// QTSimAM_CNN_LSTM_Model_6511170421082
// MI455X (gfx1250) — matrix operand order repaired and checked by reading, not compiled, not run on hardware
//
#include <hip/hip_runtime.h>

// QTSimAM_CNN_LSTM for MI455X (gfx1250, wave32, WMMA).
// Pipeline: queue stats -> f16 weight prep -> 3x (conv-WMMA + SimAM) ->
// fused simam3+transpose (f16) -> batch-sharded WMMA LSTM (16 WGs, w_ih in LDS,
// async global->LDS staging) -> classifier.

#define EPSF 1e-6f
#define ELAM 1e-4f

// Flip to 0 if the assembler rejects the async mnemonic.
#define USE_ASYNC_LDS 1

typedef __attribute__((ext_vector_type(16))) _Float16 v16h;
typedef __attribute__((ext_vector_type(8)))  _Float16 v8h;
typedef __attribute__((ext_vector_type(8)))  float    v8f;

__device__ __forceinline__ float sigf(float x) { return 1.f / (1.f + __expf(-x)); }

// ---- CDNA5 WMMA f16 16x16x32 fragment layout (wave32), ISA 7.12.2 ----
// A (16x32 row-major src): lane L holds row M=L&15; two contiguous 8-half runs
//   at k = ((L>>4)<<3) + {0..7} and +16.
// B (32x16, src transposed as Bt[n][k]): lane L holds col N=L&15;
//   two 8-half runs at k = ((L>>4)<<3) + {0..7} and +16  (two b128 loads).
// C/D (16x16 f32): lane L col N=L&15, VGPR r row M = r + (L>=16 ? 8 : 0).
template <typename P>
__device__ __forceinline__ v16h frag2(P p0, P p1) {
  v8h lo = *(const v8h*)p0;
  v8h hi = *(const v8h*)p1;
  return __builtin_shufflevector(lo, hi, 0, 1, 2, 3, 4, 5, 6, 7, 8, 9, 10, 11,
                                 12, 13, 14, 15);
}

__device__ __forceinline__ v8f wmma16(v16h a, v16h b, v8f c) {
  return __builtin_amdgcn_wmma_f32_16x16x32_f16(false, a, false, b, (short)0, c,
                                                false, false);
}

// ===================== queue layer: raw Wq/Lq + block stats =====================
__global__ __launch_bounds__(512) void queue_kernel(
    const float* __restrict__ x, const float* __restrict__ ksp,
    const float* __restrict__ kap, float* __restrict__ wq, float* __restrict__ lq,
    float* __restrict__ partials, float* __restrict__ wsum, float* __restrict__ lsum,
    int S, int F) {
  const int b = blockIdx.x, s = threadIdx.x;
  const float k_s = ksp[0], k_a = kap[0];
  const float* xe = x + ((long)b * S + s) * F;
  const float dist = xe[F - 3], air = xe[F - 5];
  const float ES  = k_s * dist + EPSF;
  const float lam = k_a / (air + EPSF);
  const float rho = fminf(lam * ES, 0.99f);
  const float W   = rho / (1.f - rho + EPSF) * ES;
  const float L   = lam * W;
  wq[(long)b * S + s] = W;
  lq[(long)b * S + s] = L;

  __shared__ float red[16 * 6];
  float wmn = W, wmx = W, lmn = L, lmx = L, ws_ = W, ls_ = L;
#pragma unroll
  for (int off = 16; off > 0; off >>= 1) {
    wmn = fminf(wmn, __shfl_down(wmn, off));
    wmx = fmaxf(wmx, __shfl_down(wmx, off));
    lmn = fminf(lmn, __shfl_down(lmn, off));
    lmx = fmaxf(lmx, __shfl_down(lmx, off));
    ws_ += __shfl_down(ws_, off);
    ls_ += __shfl_down(ls_, off);
  }
  const int wid = threadIdx.x >> 5;
  if ((threadIdx.x & 31) == 0) {
    red[wid] = wmn; red[16 + wid] = wmx; red[32 + wid] = lmn;
    red[48 + wid] = lmx; red[64 + wid] = ws_; red[80 + wid] = ls_;
  }
  __syncthreads();
  if (threadIdx.x == 0) {
    wmn = red[0]; wmx = red[16]; lmn = red[32]; lmx = red[48];
    ws_ = red[64]; ls_ = red[80];
    for (int i = 1; i < 16; i++) {
      wmn = fminf(wmn, red[i]);       wmx = fmaxf(wmx, red[16 + i]);
      lmn = fminf(lmn, red[32 + i]);  lmx = fmaxf(lmx, red[48 + i]);
      ws_ += red[64 + i];             ls_ += red[80 + i];
    }
    partials[b * 4 + 0] = wmn; partials[b * 4 + 1] = wmx;
    partials[b * 4 + 2] = lmn; partials[b * 4 + 3] = lmx;
    wsum[b] = ws_; lsum[b] = ls_;
  }
}

// ===================== global min/max + per-batch d,l =====================
__global__ __launch_bounds__(256) void finalize_kernel(
    const float* __restrict__ partials, const float* __restrict__ wsum,
    const float* __restrict__ lsum, float* __restrict__ scal,
    float* __restrict__ dvec, float* __restrict__ lvec, int S) {
  __shared__ float sh[256 * 4];
  const int t = threadIdx.x;  // == B
  sh[t]       = partials[t * 4 + 0];
  sh[256 + t] = partials[t * 4 + 1];
  sh[512 + t] = partials[t * 4 + 2];
  sh[768 + t] = partials[t * 4 + 3];
  __syncthreads();
  for (int off = 128; off > 0; off >>= 1) {
    if (t < off) {
      sh[t]       = fminf(sh[t], sh[t + off]);
      sh[256 + t] = fmaxf(sh[256 + t], sh[256 + t + off]);
      sh[512 + t] = fminf(sh[512 + t], sh[512 + t + off]);
      sh[768 + t] = fmaxf(sh[768 + t], sh[768 + t + off]);
    }
    __syncthreads();
  }
  const float wmn = sh[0], wmx = sh[256], lmn = sh[512], lmx = sh[768];
  const float winv = 1.f / (wmx - wmn + EPSF);
  const float linv = 1.f / (lmx - lmn + EPSF);
  if (t == 0) { scal[0] = wmn; scal[1] = winv; scal[2] = lmn; scal[3] = linv; }
  dvec[t] = (wsum[t] / (float)S - wmn) * winv;  // mean(Wn)
  lvec[t] = (lsum[t] / (float)S - lmn) * linv;
}

__global__ void normalize_kernel(const float* __restrict__ wq,
                                 const float* __restrict__ lq,
                                 const float* __restrict__ scal,
                                 float* __restrict__ wn, float* __restrict__ ln,
                                 int N) {
  const int i = blockIdx.x * blockDim.x + threadIdx.x;
  if (i < N) {
    wn[i] = (wq[i] - scal[0]) * scal[1];
    ln[i] = (lq[i] - scal[2]) * scal[3];
  }
}

// ===================== f32 -> f16 weight prep (with K zero-pad) =====================
__global__ void cvt_pad_f16(const float* __restrict__ src, _Float16* __restrict__ dst,
                            int scols, int dcols) {
  const long r = blockIdx.x;
  for (int c = threadIdx.x; c < dcols; c += blockDim.x)
    dst[r * dcols + c] = (c < scols) ? (_Float16)src[r * scols + c] : (_Float16)0.f;
}

// ===================== conv1d(k=3,SAME) as implicit GEMM, WMMA =====================
__global__ __launch_bounds__(256) void conv_wmma_kernel(
    const float* __restrict__ zin, long zbs, long zcs, long zss,
    const _Float16* __restrict__ w16, int kpad, int kreal,
    const float* __restrict__ bias, float* __restrict__ yout, int S, int Cout) {
  __shared__ __align__(16) _Float16 bstage[16 * 384];
  const int tid = threadIdx.x, lane = tid & 31, wv = tid >> 5;
  const int nwaves = blockDim.x >> 5;
  const int stile = blockIdx.x, b = blockIdx.z;
  const int ctile = blockIdx.y * nwaves + wv;
  const int s0 = stile << 4;
  const float* zb = zin + (long)b * zbs;
  // stage Bt[n][k] = zin[b][k/3][s0 + n + (k%3) - 1]
  for (int n = 0; n < 16; n++) {
    for (int k = tid; k < kpad; k += blockDim.x) {
      float v = 0.f;
      if (k < kreal) {
        const int ci = k / 3, dk = k - ci * 3;
        const int s = s0 + n + dk - 1;
        if (s >= 0 && s < S) v = zb[(long)ci * zcs + (long)s * zss];
      }
      bstage[n * kpad + k] = (_Float16)v;
    }
  }
  __syncthreads();

  const int mrow = (ctile << 4) + (lane & 15);
  const int ncol = lane & 15;
  const _Float16* arow = w16 + (long)mrow * kpad;
  const _Float16* brow = bstage + ncol * kpad;
  const int aoff = (lane >> 4) << 3;
  const int boff = (lane >> 4) << 3;
  v8f acc = {};
  for (int k0 = 0; k0 < kpad; k0 += 32) {
    __builtin_prefetch(arow + k0 + 32, 0, 1);  // global_prefetch_b8
    v16h a  = frag2(arow + k0 + aoff, arow + k0 + aoff + 16);
    v16h bf = frag2(brow + k0 + boff, brow + k0 + boff + 16);
    acc = wmma16(a, bf, acc);
  }
  const int hi8 = (lane >> 4) << 3;
#pragma unroll
  for (int r = 0; r < 8; r++) {
    const int c = (ctile << 4) + r + hi8;
    const float y = fmaxf(acc[r] + bias[c], 0.f);
    yout[(long)b * Cout * S + (long)c * S + (s0 + ncol)] = y;
  }
}

// ===================== SimAM scale computation (per b,c row over S) =====================
__device__ __forceinline__ float simam_scale_of_row(const float* row, int S,
                                                    float db, float lb) {
  const float v0 = row[threadIdx.x], v1 = row[threadIdx.x + 256];
  float sum = v0 + v1, sq = v0 * v0 + v1 * v1;
#pragma unroll
  for (int off = 16; off > 0; off >>= 1) {
    sum += __shfl_down(sum, off);
    sq  += __shfl_down(sq, off);
  }
  __shared__ float s_sum[8], s_sq[8];
  const int wid = threadIdx.x >> 5;
  if ((threadIdx.x & 31) == 0) { s_sum[wid] = sum; s_sq[wid] = sq; }
  __syncthreads();
  if (threadIdx.x == 0) {
    float ts = 0.f, tq = 0.f;
    for (int i = 0; i < 8; i++) { ts += s_sum[i]; tq += s_sq[i]; }
    s_sum[0] = ts; s_sq[0] = tq;
  }
  __syncthreads();
  const float mu  = s_sum[0] / (float)S;
  const float var = s_sq[0] / (float)S - mu * mu;
  return sigf(var + db + 0.5f * lb + ELAM);
}

__global__ __launch_bounds__(256) void simam_kernel(float* __restrict__ y, int S, int C,
                                                    const float* __restrict__ dvec,
                                                    const float* __restrict__ lvec) {
  const int bc = blockIdx.x;
  const int b = bc / C;
  float* row = y + (long)bc * S;  // S == 512, blockDim == 256
  const float sc = simam_scale_of_row(row, S, dvec[b], lvec[b]);
  row[threadIdx.x]       *= sc;
  row[threadIdx.x + 256] *= sc;
}

__global__ __launch_bounds__(256) void simam_stat_kernel(
    const float* __restrict__ y, int S, int C, const float* __restrict__ dvec,
    const float* __restrict__ lvec, float* __restrict__ scales) {
  const int bc = blockIdx.x;
  const int b = bc / C;
  const float sc = simam_scale_of_row(y + (long)bc * S, S, dvec[b], lvec[b]);
  if (threadIdx.x == 0) scales[bc] = sc;
}

// ===================== fused simam3-scale + transpose to f16 [b][s][c] =====================
__global__ __launch_bounds__(256) void simam_transpose_kernel(
    const float* __restrict__ src, const float* __restrict__ scales,
    _Float16* __restrict__ dst, int C, int S) {
  __shared__ float tile[16][17];
  const int b = blockIdx.z;
  const int c0 = blockIdx.y << 4, s0 = blockIdx.x << 4;
  const int tx = threadIdx.x & 15, ty = threadIdx.x >> 4;
  const float sc = scales[b * C + c0 + ty];
  tile[ty][tx] = src[((long)b * C + c0 + ty) * S + s0 + tx] * sc;
  __syncthreads();
  dst[((long)b * S + s0 + ty) * C + c0 + tx] = (_Float16)tile[tx][ty];
}

// ===================== batch-sharded Mogrifier-LSTM =====================
// Grid = B/16 independent workgroups (the recurrence never mixes batch rows).
// 256 threads = 8 waves; wave w owns hid-tile w. LDS: w_ih pinned (256KB,
// staged once with async global->LDS b128) + gin (16x160) + ms (16x256).
__global__ __launch_bounds__(256) void lstm_wmma_kernel(
    const _Float16* __restrict__ zt,  // [B][S][256] f16 (simam-scaled, transposed)
    const float* __restrict__ wn, const float* __restrict__ ln,
    const _Float16* __restrict__ modw, const float* __restrict__ modb,
    const _Float16* __restrict__ wih, const _Float16* __restrict__ whh,
    const float* __restrict__ bih, const float* __restrict__ bhh,
    float* __restrict__ hfin, int S) {
  extern __shared__ __align__(16) _Float16 lds[];
  _Float16* wihL = lds;                        // 512 x 256
  _Float16* gin  = lds + 512 * 256;            // 16 x 160
  _Float16* ms   = gin + 16 * 160;             // 16 x 256
  const int tid = threadIdx.x, lane = tid & 31, wv = tid >> 5;
  const int b0 = blockIdx.x << 4;
  const int aoff = (lane >> 4) << 3;
  const int boff = (lane >> 4) << 3;
  const int hi8 = (lane >> 4) << 3;

  // ---- one-time stage of w_ih into LDS (16384 x b128 chunks) ----
  {
    const int chunks = 512 * 256 / 8;
#if USE_ASYNC_LDS
    for (int i = tid; i < chunks; i += 256) {
      const unsigned lds_off =
          (unsigned)(unsigned long long)(wihL + (long)i * 8);  // LDS byte offset
      const unsigned long long ga = (unsigned long long)(wih + (long)i * 8);
      asm volatile("global_load_async_to_lds_b128 %0, %1, off"
                   :: "v"(lds_off), "v"(ga)
                   : "memory");
    }
    asm volatile("s_wait_asynccnt 0x0" ::: "memory");
#else
    for (int i = tid; i < chunks; i += 256)
      *(v8h*)(wihL + (long)i * 8) = *(const v8h*)(wih + (long)i * 8);
#endif
  }
  for (int i = tid; i < 16 * 160; i += 256) gin[i] = (_Float16)0.f;
  float creg[8], hreg[8];
#pragma unroll
  for (int r = 0; r < 8; r++) { creg[r] = 0.f; hreg[r] = 0.f; }
  __syncthreads();

  for (int t = 0; t < S; ++t) {
    if (tid < 16) {  // d_t, l_t into gate-input cols 128,129
      gin[tid * 160 + 128] = (_Float16)wn[(long)(b0 + tid) * S + t];
      gin[tid * 160 + 129] = (_Float16)ln[(long)(b0 + tid) * S + t];
    }
    __syncthreads();

    // ---- m = sigmoid([h,d,l] @ modw^T + modb); ms = m .* s_t ----
#pragma unroll
    for (int tn = 0; tn < 2; ++tn) {
      const int jt = wv * 2 + tn;                 // 16 j-tiles over 8 waves
      const int jcol = (jt << 4) + (lane & 15);
      const _Float16* arow = gin + (lane & 15) * 160;
      const _Float16* bte  = modw + (long)jcol * 160;
      v8f acc = {};
      for (int k0 = 0; k0 < 160; k0 += 32) {
        v16h a  = frag2(arow + k0 + aoff, arow + k0 + aoff + 16);
        v16h bf = frag2(bte + k0 + boff, bte + k0 + boff + 16);
        acc = wmma16(a, bf, acc);
      }
      const float mb = modb[jcol];
#pragma unroll
      for (int r = 0; r < 8; r++) {
        const int bb = r + hi8;                   // local batch row 0..15
        const float m  = sigf(acc[r] + mb);
        const float sv = (float)zt[((long)(b0 + bb) * S + t) * 256 + jcol];
        ms[bb * 256 + jcol] = (_Float16)(m * sv);
      }
    }
    __syncthreads();

    // ---- g = ms @ wih^T + h @ whh^T + biases; LSTM cell update ----
    {
      const int ncol = (wv << 4) + (lane & 15);   // hid 0..127, wave owns tile wv
      const _Float16* msrow = ms + (lane & 15) * 256;
      const _Float16* hrow  = gin + (lane & 15) * 160;
      v8f acc4[4] = {};
      for (int k0 = 0; k0 < 256; k0 += 32) {
        v16h a = frag2(msrow + k0 + aoff, msrow + k0 + aoff + 16);
#pragma unroll
        for (int q = 0; q < 4; q++) {
          const _Float16* bte = wihL + (long)(q * 128 + ncol) * 256;  // LDS
          v16h bf = frag2(bte + k0 + boff, bte + k0 + boff + 16);
          acc4[q] = wmma16(a, bf, acc4[q]);
        }
      }
      for (int k0 = 0; k0 < 128; k0 += 32) {
        v16h a = frag2(hrow + k0 + aoff, hrow + k0 + aoff + 16);
#pragma unroll
        for (int q = 0; q < 4; q++) {
          const _Float16* bte = whh + (long)(q * 128 + ncol) * 128;
          v16h bf = frag2(bte + k0 + boff, bte + k0 + boff + 16);
          acc4[q] = wmma16(a, bf, acc4[q]);
        }
      }
      const float bI = bih[ncol] + bhh[ncol];
      const float bF = bih[128 + ncol] + bhh[128 + ncol];
      const float bG = bih[256 + ncol] + bhh[256 + ncol];
      const float bO = bih[384 + ncol] + bhh[384 + ncol];
#pragma unroll
      for (int r = 0; r < 8; r++) {
        const float iv = sigf(acc4[0][r] + bI);
        const float fv = sigf(acc4[1][r] + bF);
        const float gv = tanhf(acc4[2][r] + bG);
        const float ov = sigf(acc4[3][r] + bO);
        const float c  = fv * creg[r] + iv * gv;
        creg[r] = c;
        hreg[r] = ov * tanhf(c);
      }
    }
    __syncthreads();  // all reads of gin/ms done before h overwrite
    {
      const int ncol = (wv << 4) + (lane & 15);
#pragma unroll
      for (int r = 0; r < 8; r++) {
        const int bb = r + hi8;
        gin[bb * 160 + ncol] = (_Float16)hreg[r];
        if (t == S - 1) hfin[(long)(b0 + bb) * 128 + ncol] = hreg[r];
      }
    }
    __syncthreads();
  }
}

// ===================== tiny classifier =====================
__global__ __launch_bounds__(256) void cls_kernel(const float* __restrict__ hfin,
                                                  const float* __restrict__ cw,
                                                  const float* __restrict__ cb,
                                                  float* __restrict__ out) {
  const int b = threadIdx.x;
  for (int nc = 0; nc < 8; nc++) {
    float s = cb[nc];
    for (int k = 0; k < 128; k++) s += hfin[b * 128 + k] * cw[nc * 128 + k];
    out[b * 8 + nc] = s;
  }
}

// ===================== host =====================
extern "C" void kernel_launch(void* const* d_in, const int* in_sizes, int n_in,
                              void* d_out, int out_size, void* d_ws, size_t ws_size,
                              hipStream_t stream) {
  (void)in_sizes; (void)n_in; (void)out_size; (void)ws_size;
  const int B = 256, S = 512, F = 16, C1 = 64, C2 = 128, C3 = 256;
  const float* x    = (const float*)d_in[0];
  const float* ks   = (const float*)d_in[1];
  const float* ka   = (const float*)d_in[2];
  const float* w1   = (const float*)d_in[3];
  const float* b1   = (const float*)d_in[4];
  const float* w2   = (const float*)d_in[5];
  const float* b2   = (const float*)d_in[6];
  const float* w3   = (const float*)d_in[7];
  const float* b3   = (const float*)d_in[8];
  const float* modw = (const float*)d_in[9];
  const float* modb = (const float*)d_in[10];
  const float* wih  = (const float*)d_in[11];
  const float* whh  = (const float*)d_in[12];
  const float* bih  = (const float*)d_in[13];
  const float* bhh  = (const float*)d_in[14];
  const float* clw  = (const float*)d_in[15];
  const float* clb  = (const float*)d_in[16];
  float* out = (float*)d_out;

  char* base = (char*)d_ws;
  size_t off = 0;
  auto alloc = [&](size_t bytes) -> char* {
    char* r = base + off;
    off = (off + bytes + 255) & ~(size_t)255;
    return r;
  };
  float* wq       = (float*)alloc((size_t)B * S * 4);
  float* lq       = (float*)alloc((size_t)B * S * 4);
  float* wn       = (float*)alloc((size_t)B * S * 4);
  float* ln       = (float*)alloc((size_t)B * S * 4);
  float* partials = (float*)alloc((size_t)B * 4 * 4);
  float* wsum     = (float*)alloc((size_t)B * 4);
  float* lsum     = (float*)alloc((size_t)B * 4);
  float* scal     = (float*)alloc(16 * 4);
  float* dvec     = (float*)alloc((size_t)B * 4);
  float* lvec     = (float*)alloc((size_t)B * 4);
  float* scales3  = (float*)alloc((size_t)B * C3 * 4);
  float* hfin     = (float*)alloc((size_t)B * 128 * 4);
  _Float16* w1h   = (_Float16*)alloc((size_t)C1 * 64 * 2);
  _Float16* w2h   = (_Float16*)alloc((size_t)C2 * 192 * 2);
  _Float16* w3h   = (_Float16*)alloc((size_t)C3 * 384 * 2);
  _Float16* mwh   = (_Float16*)alloc((size_t)256 * 160 * 2);
  _Float16* wihh  = (_Float16*)alloc((size_t)512 * 256 * 2);
  _Float16* whhh  = (_Float16*)alloc((size_t)512 * 128 * 2);
  float* z1       = (float*)alloc((size_t)B * C1 * S * 4);
  float* z2       = (float*)alloc((size_t)B * C2 * S * 4);
  float* z3       = (float*)alloc((size_t)B * C3 * S * 4);
  // zt (f16, B*S*C3*2 = 67MB) reuses the dead z1+z2 region (100MB) after conv3
  _Float16* zt = (_Float16*)z1;

  // --- queue layer + global stats ---
  queue_kernel<<<B, 512, 0, stream>>>(x, ks, ka, wq, lq, partials, wsum, lsum, S, F);
  finalize_kernel<<<1, 256, 0, stream>>>(partials, wsum, lsum, scal, dvec, lvec, S);
  normalize_kernel<<<(B * S) / 256, 256, 0, stream>>>(wq, lq, scal, wn, ln, B * S);

  // --- f16 weight prep (K zero-padded to multiples of 32) ---
  cvt_pad_f16<<<C1, 128, 0, stream>>>(w1, w1h, 48, 64);
  cvt_pad_f16<<<C2, 128, 0, stream>>>(w2, w2h, 192, 192);
  cvt_pad_f16<<<C3, 128, 0, stream>>>(w3, w3h, 384, 384);
  cvt_pad_f16<<<256, 128, 0, stream>>>(modw, mwh, 130, 160);
  cvt_pad_f16<<<512, 128, 0, stream>>>(wih, wihh, 256, 256);
  cvt_pad_f16<<<512, 128, 0, stream>>>(whh, whhh, 128, 128);

  // --- conv + SimAM stack ---
  {
    dim3 g(S / 16, 1, B);  // 4 waves cover C1/16 = 4 c-tiles
    conv_wmma_kernel<<<g, 128, 0, stream>>>(x, (long)S * F, 1L, (long)F, w1h, 64, 48,
                                            b1, z1, S, C1);
    simam_kernel<<<B * C1, 256, 0, stream>>>(z1, S, C1, dvec, lvec);
  }
  {
    dim3 g(S / 16, 1, B);  // 8 waves cover C2/16 = 8 c-tiles
    conv_wmma_kernel<<<g, 256, 0, stream>>>(z1, (long)C1 * S, (long)S, 1L, w2h, 192,
                                            192, b2, z2, S, C2);
    simam_kernel<<<B * C2, 256, 0, stream>>>(z2, S, C2, dvec, lvec);
  }
  {
    dim3 g(S / 16, 2, B);  // 2 x 8 waves cover C3/16 = 16 c-tiles
    conv_wmma_kernel<<<g, 256, 0, stream>>>(z2, (long)C2 * S, (long)S, 1L, w3h, 384,
                                            384, b3, z3, S, C3);
    simam_stat_kernel<<<B * C3, 256, 0, stream>>>(z3, S, C3, dvec, lvec, scales3);
  }
  {
    dim3 g(S / 16, C3 / 16, B);
    simam_transpose_kernel<<<g, 256, 0, stream>>>(z3, scales3, zt, C3, S);
  }

  // --- batch-sharded LSTM: 16 independent WGs, w_ih pinned in LDS ---
  const size_t lds_bytes =
      (size_t)(512 * 256 + 16 * 160 + 16 * 256) * sizeof(_Float16);  // ~269KB
  hipFuncSetAttribute((const void*)lstm_wmma_kernel,
                      hipFuncAttributeMaxDynamicSharedMemorySize, (int)lds_bytes);
  lstm_wmma_kernel<<<B / 16, 256, lds_bytes, stream>>>(zt, wn, ln, mwh, modb, wihh,
                                                       whhh, bih, bhh, hfin, S);

  // --- classifier ---
  cls_kernel<<<1, 256, 0, stream>>>(hfin, clw, clb, out);
}
